// StackRGGCN_68229850465014
// MI455X (gfx1250) — matrix operand order repaired and checked by reading, not compiled, not run on hardware
//
#include <hip/hip_runtime.h>
#include <hip/hip_bf16.h>
#include <math.h>

// Problem constants (from reference)
#define NUU   10000
#define NVV   40000
#define NN    50000      // NUU + NVV
#define DD    512
#define OUTD  640
#define SSUP  5
#define EED   100000
#define DSD   128        // OUTD / SSUP
#define BN_EPS 1e-3f

typedef __attribute__((ext_vector_type(16))) __bf16 v16bf;
typedef __attribute__((ext_vector_type(8)))  __bf16 v8bf;
typedef __attribute__((ext_vector_type(8)))  float  v8f;

// ---------------------------------------------------------------------------
// 1) Convert x = concat(users, items) to bf16 (51 MB -> lives in 192MB L2)
// ---------------------------------------------------------------------------
__global__ __launch_bounds__(256) void k_convert_x(
    const float* __restrict__ users, const float* __restrict__ items,
    __bf16* __restrict__ xb) {
  size_t idx = (size_t)blockIdx.x * 256 + threadIdx.x;
  const size_t total = (size_t)NN * DD;
  if (idx >= total) return;
  const size_t usz = (size_t)NUU * DD;
  float v = (idx < usz) ? users[idx] : items[idx - usz];
  xb[idx] = (__bf16)v;
}

// ---------------------------------------------------------------------------
// 2) Transpose+convert the 5 weights [D,OUT] f32 -> Wt[w][out][k] bf16
//    so B-fragments are contiguous 16B loads per lane.
// ---------------------------------------------------------------------------
__global__ __launch_bounds__(256) void k_convert_w(
    const float* __restrict__ Ui, const float* __restrict__ Uj,
    const float* __restrict__ Vi, const float* __restrict__ Vj,
    const float* __restrict__ R, __bf16* __restrict__ Wt) {
  size_t idx = (size_t)blockIdx.x * 256 + threadIdx.x;
  const size_t per = (size_t)OUTD * DD;
  if (idx >= 5 * per) return;
  int w = (int)(idx / per);
  size_t rem = idx % per;
  int n = (int)(rem / DD);
  int k = (int)(rem % DD);
  const float* W = (w == 0) ? Ui : (w == 1) ? Uj : (w == 2) ? Vi : (w == 3) ? Vj : R;
  Wt[idx] = (__bf16)W[(size_t)k * OUTD + n];
}

// ---------------------------------------------------------------------------
// 3) Zero BN accumulators + zero-bias buffer (sum[640]|sumsq[640]|zeros[640])
// ---------------------------------------------------------------------------
__global__ __launch_bounds__(256) void k_zero_stats(float* __restrict__ p) {
  int idx = blockIdx.x * 256 + threadIdx.x;
  if (idx < 3 * OUTD) p[idx] = 0.0f;
}

// ---------------------------------------------------------------------------
// WMMA fragment loaders (ISA 16-bit layouts, wave32):
//  A 16x32: lane r=L&15 is row M; per-lane K = {hi*8+0..7} U {16+hi*8+0..7}
//  B 32x16: lane r=L&15 is col N; per-lane K = {hi*8+0..7} U {16+hi*8+0..7}
// Both become aligned b128 global loads from row-major bf16 rows.
// ---------------------------------------------------------------------------
__device__ __forceinline__ v16bf load_a_frag(const __bf16* rowp, int k0, int hi) {
  const v8bf c0 = *reinterpret_cast<const v8bf*>(rowp + k0 + hi * 8);
  const v8bf c1 = *reinterpret_cast<const v8bf*>(rowp + k0 + 16 + hi * 8);
  v16bf out;
#pragma unroll
  for (int i = 0; i < 8; ++i) { out[i] = c0[i]; out[i + 8] = c1[i]; }
  return out;
}

__device__ __forceinline__ v16bf load_b_frag(const __bf16* rowp, int k0, int hi) {
  const v8bf c0 = *reinterpret_cast<const v8bf*>(rowp + k0 + hi * 8);
  const v8bf c1 = *reinterpret_cast<const v8bf*>(rowp + k0 + 16 + hi * 8);
  v16bf out;
#pragma unroll
  for (int i = 0; i < 8; ++i) { out[i] = c0[i]; out[i + 8] = c1[i]; }
  return out;
}

// ---------------------------------------------------------------------------
// 4) Fused 5-way GEMM: C_z[N,640] = xb[N,512] @ W_z, z = blockIdx.z
//    z0 -> H   (+bu folded)   z1 -> PUj
//    z2 -> PVi (+bv folded)   z3 -> PVj   z4 -> d_out (x@R)
//    Block = 256 thr = 8 waves; block tile 128x128; wave tile 32x64
//    (2x4 wmma accumulators of 16x16, K-loop step 32).
//    bias is ALWAYS a valid pointer (zeros buffer when unused) -> no branches.
// ---------------------------------------------------------------------------
__global__ __launch_bounds__(256) void k_gemm(
    const __bf16* __restrict__ xb, const __bf16* __restrict__ Wt,
    float* __restrict__ H, float* __restrict__ PUj,
    float* __restrict__ PVi, float* __restrict__ PVj, float* __restrict__ XR,
    const float* __restrict__ bu, const float* __restrict__ bv,
    const float* __restrict__ zeros) {
  const int z = blockIdx.z;
  const __bf16* W = Wt + (size_t)z * OUTD * DD;
  float* Cout;
  const float* bias;
  switch (z) {
    case 0: Cout = H;   bias = bu;    break;
    case 1: Cout = PUj; bias = zeros; break;
    case 2: Cout = PVi; bias = bv;    break;
    case 3: Cout = PVj; bias = zeros; break;
    default: Cout = XR; bias = zeros; break;
  }

  const int lane = threadIdx.x & 31;
  const int wave = threadIdx.x >> 5;      // 0..7
  const int r    = lane & 15;
  const int hi   = lane >> 4;
  const int waveM = wave >> 1;            // 0..3 -> 32-row slices
  const int waveN = wave & 1;             // 0..1 -> 64-col slices
  const int rowBase = blockIdx.x * 128 + waveM * 32;
  const int colBase = blockIdx.y * 128 + waveN * 64;

  // A row pointers (clamp rows in the final partial tile; stores are guarded
  // only on the slow path below)
  const __bf16* arow[2];
#pragma unroll
  for (int mi = 0; mi < 2; ++mi) {
    int rr = rowBase + mi * 16 + r;
    if (rr >= NN) rr = NN - 1;
    arow[mi] = xb + (size_t)rr * DD;
  }
  // B row pointers (OUTD % 128 == 0, always in range)
  const __bf16* brow[4];
#pragma unroll
  for (int nj = 0; nj < 4; ++nj)
    brow[nj] = W + (size_t)(colBase + nj * 16 + r) * DD;

  v8f acc[2][4] = {};

#pragma unroll 2
  for (int k0 = 0; k0 < DD; k0 += 32) {
    v16bf a[2], b[4];
#pragma unroll
    for (int mi = 0; mi < 2; ++mi) a[mi] = load_a_frag(arow[mi], k0, hi);
#pragma unroll
    for (int nj = 0; nj < 4; ++nj) b[nj] = load_b_frag(brow[nj], k0, hi);
#pragma unroll
    for (int mi = 0; mi < 2; ++mi)
#pragma unroll
      for (int nj = 0; nj < 4; ++nj)
        acc[mi][nj] = __builtin_amdgcn_wmma_f32_16x16x32_bf16(
            false, a[mi], false, b[nj], (short)0, acc[mi][nj], false, false);
  }

  // Per-lane bias for each column subtile (unconditional, branch-free)
  float badd[4];
#pragma unroll
  for (int nj = 0; nj < 4; ++nj) badd[nj] = bias[colBase + nj * 16 + r];

  // Store: C/D layout -> lane element v is (row = v + hi*8, col = r) in tile.
  // Wave-uniform fast path: whole 32-row wave tile in range (all but the last
  // partial row tile take this branch -> zero exec-mask churn).
  if (rowBase + 32 <= NN) {
#pragma unroll
    for (int mi = 0; mi < 2; ++mi) {
      const size_t rb = (size_t)(rowBase + mi * 16 + hi * 8) * OUTD;
#pragma unroll
      for (int nj = 0; nj < 4; ++nj) {
        const int col = colBase + nj * 16 + r;
#pragma unroll
        for (int v = 0; v < 8; ++v)
          Cout[rb + (size_t)v * OUTD + col] = acc[mi][nj][v] + badd[nj];
      }
    }
  } else {
#pragma unroll
    for (int mi = 0; mi < 2; ++mi) {
#pragma unroll
      for (int nj = 0; nj < 4; ++nj) {
        const int col = colBase + nj * 16 + r;
#pragma unroll
        for (int v = 0; v < 8; ++v) {
          const int row = rowBase + mi * 16 + hi * 8 + v;
          if (row < NN)
            Cout[(size_t)row * OUTD + col] = acc[mi][nj][v] + badd[nj];
        }
      }
    }
  }
}

// ---------------------------------------------------------------------------
// 5) Edge kernel: gate = sigmoid(Vix[ee]+Vjx[es]+bv) (bv folded into PVi);
//    H[ee] += gate * Ujx[es] via global f32 atomics (avg 2 edges/node).
//    128 lanes per edge -> fully coalesced 512B rows per gather.
// ---------------------------------------------------------------------------
__global__ __launch_bounds__(256) void k_edges(
    const float* __restrict__ PVi, const float* __restrict__ PVj,
    const float* __restrict__ PUj, float* __restrict__ H,
    const int* __restrict__ e_start, const int* __restrict__ e_end) {
  const int s = blockIdx.y;
  const int t = blockIdx.x * 256 + threadIdx.x;  // gridDim.x*256 == EED*DSD
  const int e = t >> 7;                          // edge within support
  const int k = t & 127;
  const int es = e_start[s * EED + e];
  const int ee = e_end[s * EED + e];
  const int c = s * DSD + k;
  float g = PVi[(size_t)ee * OUTD + c] + PVj[(size_t)es * OUTD + c];
  g = 1.0f / (1.0f + __expf(-g));
  const float msg = g * PUj[(size_t)es * OUTD + c];
  __hip_atomic_fetch_add(&H[(size_t)ee * OUTD + c], msg,
                         __ATOMIC_RELAXED, __HIP_MEMORY_SCOPE_AGENT);
}

// ---------------------------------------------------------------------------
// 6) BN partial stats: block = (64 channels x 4 row-threads), grid (10,16).
//    LDS tree-reduce then one atomic per channel per block.
// ---------------------------------------------------------------------------
__global__ __launch_bounds__(256) void k_bnstats(
    const float* __restrict__ H, float* __restrict__ sum,
    float* __restrict__ sumsq) {
  __shared__ float s_s[256];
  __shared__ float s_q[256];
  const int c  = blockIdx.x * 64 + (threadIdx.x & 63);
  const int rg = threadIdx.x >> 6;        // 0..3
  const int rowsPer = NN / 16;            // 3125 rows per blockIdx.y
  const int row0 = blockIdx.y * rowsPer;
  float s = 0.0f, q = 0.0f;
  for (int i = row0 + rg; i < row0 + rowsPer; i += 4) {
    const float v = H[(size_t)i * OUTD + c];
    s += v; q += v * v;
  }
  s_s[threadIdx.x] = s; s_q[threadIdx.x] = q;
  __syncthreads();
  if (threadIdx.x < 64) {
    const float ts = s_s[threadIdx.x] + s_s[threadIdx.x + 64] +
                     s_s[threadIdx.x + 128] + s_s[threadIdx.x + 192];
    const float tq = s_q[threadIdx.x] + s_q[threadIdx.x + 64] +
                     s_q[threadIdx.x + 128] + s_q[threadIdx.x + 192];
    __hip_atomic_fetch_add(&sum[c], ts, __ATOMIC_RELAXED, __HIP_MEMORY_SCOPE_AGENT);
    __hip_atomic_fetch_add(&sumsq[c], tq, __ATOMIC_RELAXED, __HIP_MEMORY_SCOPE_AGENT);
  }
}

// ---------------------------------------------------------------------------
// 7) Finalize: out = relu( relu(BN(h)*gamma+beta) + x@R )  (x@R already in out)
// ---------------------------------------------------------------------------
__global__ __launch_bounds__(256) void k_finalize(
    const float* __restrict__ H, const float* __restrict__ sum,
    const float* __restrict__ sumsq, const float* __restrict__ gamma,
    const float* __restrict__ beta, float* __restrict__ out) {
  const size_t idx = (size_t)blockIdx.x * 256 + threadIdx.x;
  if (idx >= (size_t)NN * OUTD) return;
  const int c = (int)(idx % OUTD);
  const float inv_n = 1.0f / (float)NN;
  const float mu = sum[c] * inv_n;
  const float var = sumsq[c] * inv_n - mu * mu;
  const float rstd = rsqrtf(var + BN_EPS);
  float bn = (H[idx] - mu) * rstd * gamma[c] + beta[c];
  bn = fmaxf(bn, 0.0f);
  out[idx] = fmaxf(bn + out[idx], 0.0f);
}

// ---------------------------------------------------------------------------
extern "C" void kernel_launch(void* const* d_in, const int* in_sizes, int n_in,
                              void* d_out, int out_size, void* d_ws, size_t ws_size,
                              hipStream_t stream) {
  const float* users = (const float*)d_in[0];
  const float* items = (const float*)d_in[1];
  const float* Ui    = (const float*)d_in[2];
  const float* Uj    = (const float*)d_in[3];
  const float* Vi    = (const float*)d_in[4];
  const float* Vj    = (const float*)d_in[5];
  const float* bu    = (const float*)d_in[6];
  const float* bv    = (const float*)d_in[7];
  const float* R     = (const float*)d_in[8];
  const float* gamma = (const float*)d_in[9];
  const float* beta  = (const float*)d_in[10];
  const int* e_start = (const int*)d_in[11];
  const int* e_end   = (const int*)d_in[12];
  float* out = (float*)d_out;

  // Workspace carve-up (~566 MB)
  char* w = (char*)d_ws;
  __bf16* xb = (__bf16*)w; w += (size_t)NN * DD * sizeof(__bf16);
  __bf16* Wt = (__bf16*)w; w += (size_t)5 * OUTD * DD * sizeof(__bf16);
  float* H    = (float*)w; w += (size_t)NN * OUTD * sizeof(float);
  float* PUj  = (float*)w; w += (size_t)NN * OUTD * sizeof(float);
  float* PVi  = (float*)w; w += (size_t)NN * OUTD * sizeof(float);
  float* PVj  = (float*)w; w += (size_t)NN * OUTD * sizeof(float);
  float* sum  = (float*)w; w += (size_t)OUTD * sizeof(float);
  float* sumsq= (float*)w; w += (size_t)OUTD * sizeof(float);
  float* zeros= (float*)w; w += (size_t)OUTD * sizeof(float);

  // 1) x -> bf16
  k_convert_x<<<(NN * (size_t)DD + 255) / 256, 256, 0, stream>>>(users, items, xb);
  // 2) weights -> transposed bf16
  k_convert_w<<<(5 * (size_t)OUTD * DD + 255) / 256, 256, 0, stream>>>(Ui, Uj, Vi, Vj, R, Wt);
  // 3) zero BN accumulators + zero-bias buffer (sum|sumsq|zeros contiguous)
  k_zero_stats<<<(3 * OUTD + 255) / 256, 256, 0, stream>>>(sum);
  // 4) five GEMMs via WMMA; z picks weight/output/bias
  dim3 ggrid((NN + 127) / 128, OUTD / 128, 5);
  k_gemm<<<ggrid, 256, 0, stream>>>(xb, Wt, H, PUj, PVi, PVj, out, bu, bv, zeros);
  // 5) edge gate + scatter into H
  dim3 egrid((EED * DSD) / 256, SSUP, 1);
  k_edges<<<egrid, 256, 0, stream>>>(PVi, PVj, PUj, H, e_start, e_end);
  // 6) BN stats
  dim3 bgrid(OUTD / 64, 16, 1);
  k_bnstats<<<bgrid, 256, 0, stream>>>(H, sum, sumsq);
  // 7) finalize into d_out
  k_finalize<<<((size_t)NN * OUTD + 255) / 256, 256, 0, stream>>>(H, sum, sumsq, gamma, beta, out);
}
